// Attention_80504866996406
// MI455X (gfx1250) — matrix operand order repaired and checked by reading, not compiled, not run on hardware
//
#include <hip/hip_runtime.h>
#include <hip/hip_bf16.h>

// ---------------------------------------------------------------------------
// Transformer block for MI455X (gfx1250, wave32, WMMA + Tensor Data Mover).
// All GEMMs run through v_wmma_f32_16x16x32_bf16 (bf16 in, f32 accum).
// GEMM staging uses TENSOR_LOAD_TO_LDS (TDM) double-buffered when available.
// Wave tile 64x64 (16 WMMA / 16 ds_load_b128 per K-step) -> 2x FLOP per LDS
// byte vs 64x32, halving barrier and TDM-descriptor rate per FLOP.
// ---------------------------------------------------------------------------

typedef __bf16 bf16_t;
typedef __attribute__((ext_vector_type(16))) __bf16 v16bf;
typedef __attribute__((ext_vector_type(8)))  __bf16 v8bf;
typedef __attribute__((ext_vector_type(8)))  float  v8f;
typedef __attribute__((ext_vector_type(4)))  unsigned int v4u;
typedef __attribute__((ext_vector_type(8)))  int v8i;
typedef __attribute__((ext_vector_type(4)))  int v4i;

#define BB 8
#define SS 2048
#define EE 1024
#define HH 4096
#define LN_EPS 1e-5f

#if defined(__gfx1250__) && __has_builtin(__builtin_amdgcn_tensor_load_to_lds)
#define USE_TDM 1
#else
#define USE_TDM 0
#endif

#if USE_TDM
// Issue one TDM descriptor: load a 32 x rows element bf16 tile (rows of 64
// data bytes, row stride `ld` elements) from global into LDS at lds_addr,
// inserting 16B of padding after every 64B (-> 80B LDS rows == [.][40]).
__device__ __forceinline__ void tdm_load_tile32(const bf16_t* g,
                                                unsigned lds_addr,
                                                unsigned ld, unsigned rows)
{
    const unsigned long long ga = (unsigned long long)(size_t)g;
    v4u g0;
    g0.x = 1u;                                    // count=1 (valid user D#)
    g0.y = lds_addr;                              // LDS byte address
    g0.z = (unsigned)ga;                          // global_addr[31:0]
    g0.w = (unsigned)((ga >> 32) & 0x01FFFFFFu)   // global_addr[56:32]
         | (2u << 30);                            // type=2 ("image")
    v8i g1;
    g1[0] = (int)((1u << 16)     // data_size = 2 bytes
                | (1u << 20)     // pad_enable
                | (3u << 22)     // pad_interval: 16 DWORDs (64B)
                | (3u << 25));   // pad_amount:   4 DWORDs (16B)
    g1[1] = (int)(32u << 16);                     // tensor_dim0 = 32 elems
    g1[2] = (int)(rows << 16);                    // tensor_dim1 = rows
    g1[3] = (int)(32u << 16);                     // tile_dim0 = 32
    g1[4] = (int)rows;                            // tile_dim1 = rows, tile_dim2=0
    g1[5] = (int)ld;                              // tensor_dim0_stride[31:0]
    g1[6] = 0;                                    // stride hi / dim1_stride
    g1[7] = 0;
    const v4i z4 = {0, 0, 0, 0};
#if __clang_major__ >= 23
    const v8i z8 = {0, 0, 0, 0, 0, 0, 0, 0};
    __builtin_amdgcn_tensor_load_to_lds(g0, g1, z4, z4, z8, 0);
#else
    __builtin_amdgcn_tensor_load_to_lds(g0, g1, z4, z4, 0);
#endif
}
#endif

// Shared fragment-load + 16x WMMA for one 32-deep K step (64x64 wave tile).
__device__ __forceinline__ void compute_tile32(
    const bf16_t (*__restrict__ pA)[40], const bf16_t (*__restrict__ pB)[40],
    v8f (&acc)[4][4], int wm, int wn, int l15, int akb, int bkb)
{
    union Frag { v16bf v; v8bf h[2]; };
    Frag af[4], bfg[4];
    // A fragment (16x32): lane holds row M=l15, K = {akb..+7, 16+akb..+7}
#pragma unroll
    for (int mf = 0; mf < 4; ++mf) {
        const int r = wm + mf * 16 + l15;
        af[mf].h[0] = *(const v8bf*)&pA[r][akb];
        af[mf].h[1] = *(const v8bf*)&pA[r][16 + akb];
    }
    // B fragment (32x16): lane holds col N=l15, K = {akb..+7, 16+akb..+7}
#pragma unroll
    for (int nf = 0; nf < 4; ++nf) {
        const int c = wn + nf * 16 + l15;
        bfg[nf].h[0] = *(const v8bf*)&pB[c][akb];
        bfg[nf].h[1] = *(const v8bf*)&pB[c][16 + akb];
    }
#pragma unroll
    for (int mf = 0; mf < 4; ++mf)
#pragma unroll
        for (int nf = 0; nf < 4; ++nf)
            acc[mf][nf] = __builtin_amdgcn_wmma_f32_16x16x32_bf16(
                false, af[mf].v, false, bfg[nf].v,
                (short)0, acc[mf][nf], false, false);
}

// ---------------------------------------------------------------------------
// Core bf16 WMMA GEMM:  D[M,N] = A[M,K] * Bt[N,K]^T  (Bt is N-major)
// Block: 256 threads = 8 waves. Workgroup tile 128x256, wave grid 2(M)x4(N),
// wave tile 64x64 (4x4 grid of 16x16 WMMA frags), K-step 32.
// EPI: 0 = store f32
//      1 = +bias, store bf16
//      2 = +bias, store f32
//      3 = +bias, relu, store bf16
//      4 = +bias, relu, +residual, store f32
// ---------------------------------------------------------------------------
template <int EPI>
__global__ __launch_bounds__(256) void gemm_bf16_wmma(
    const bf16_t* __restrict__ A, int lda, size_t batchA,
    const bf16_t* __restrict__ Bt, int ldb, size_t batchB,
    float* __restrict__ Df, bf16_t* __restrict__ Db, int ldd, size_t batchD,
    const float* __restrict__ bias,
    const float* __restrict__ resid, size_t batchR,
    int K)
{
    const int tid  = threadIdx.x;
    const int lane = tid & 31;
    const int wave = tid >> 5;
    const int wm   = (wave >> 2) * 64;   // wave M offset: 0 or 64
    const int wn   = (wave & 3) * 64;    // wave N offset: 0,64,128,192
    const int l15  = lane & 15;
    const int hi   = lane >> 4;          // 0 for lanes 0-15, 1 for 16-31
    const int akb  = hi * 8;             // A frag K base (split pattern)
    const int bkb  = hi * 16;            // unused: B frag uses akb (split) like A

    const size_t tileM = (size_t)blockIdx.y * 128;
    const size_t tileN = (size_t)blockIdx.x * 256;
    const size_t bz    = blockIdx.z;

    A  += bz * batchA + tileM * (size_t)lda;
    Bt += bz * batchB + tileN * (size_t)ldb;

    const v8f vzero = {0.f, 0.f, 0.f, 0.f, 0.f, 0.f, 0.f, 0.f};
    v8f acc[4][4];
#pragma unroll
    for (int i = 0; i < 4; ++i)
#pragma unroll
        for (int j = 0; j < 4; ++j) acc[i][j] = vzero;

#if USE_TDM
    // ---- TDM double-buffered staging: DMA tile k+1 overlaps WMMA on tile k.
    __shared__ bf16_t sA[2][128][40];    // 80B rows: 64B data + 16B TDM pad
    __shared__ bf16_t sB[2][256][40];
    const unsigned ldsA0 = (unsigned)(size_t)(void*)&sA[0][0][0];
    const unsigned ldsA1 = (unsigned)(size_t)(void*)&sA[1][0][0];
    const unsigned ldsB0 = (unsigned)(size_t)(void*)&sB[0][0][0];
    const unsigned ldsB1 = (unsigned)(size_t)(void*)&sB[1][0][0];

    const int nk = K / 32;
    if (wave == 0) {
        tdm_load_tile32(A, ldsA0, (unsigned)lda, 128u);
        tdm_load_tile32(Bt, ldsB0, (unsigned)ldb, 256u);
    }
    for (int kt = 0; kt < nk; ++kt) {
        const int cur = kt & 1;
        __builtin_amdgcn_s_wait_tensorcnt(0);  // wave0: tile kt landed in LDS
        __syncthreads();                       // all see tile kt; buf cur^1 free
        if (wave == 0 && kt + 1 < nk) {        // start DMA for tile kt+1
            tdm_load_tile32(A  + (size_t)(kt + 1) * 32,
                            cur ? ldsA0 : ldsA1, (unsigned)lda, 128u);
            tdm_load_tile32(Bt + (size_t)(kt + 1) * 32,
                            cur ? ldsB0 : ldsB1, (unsigned)ldb, 256u);
        }
        compute_tile32(sA[cur], sB[cur], acc, wm, wn, l15, akb, bkb);
    }
#else
    // ---- Fallback: manual global->LDS staging (single buffer).
    __shared__ bf16_t sA[128][40];
    __shared__ bf16_t sB[256][40];
    const int crow = tid >> 2;          // 0..63
    const int ccol = (tid & 3) * 8;     // 0,8,16,24
    for (int k0 = 0; k0 < K; k0 += 32) {
        const bf16_t* Ag = A  + crow * (size_t)lda + k0 + ccol;
        const bf16_t* Bg = Bt + crow * (size_t)ldb + k0 + ccol;
        v8bf a0 = *(const v8bf*)Ag;
        v8bf a1 = *(const v8bf*)(Ag + (size_t)64 * lda);
        v8bf b0 = *(const v8bf*)Bg;
        v8bf b1 = *(const v8bf*)(Bg + (size_t)64 * ldb);
        v8bf b2 = *(const v8bf*)(Bg + (size_t)128 * ldb);
        v8bf b3 = *(const v8bf*)(Bg + (size_t)192 * ldb);
        if (k0 + 32 < K) {
            __builtin_prefetch(Ag + 32, 0, 3);
            __builtin_prefetch(Bg + 32, 0, 3);
        }
        *(v8bf*)&sA[crow][ccol]       = a0;
        *(v8bf*)&sA[crow + 64][ccol]  = a1;
        *(v8bf*)&sB[crow][ccol]       = b0;
        *(v8bf*)&sB[crow + 64][ccol]  = b1;
        *(v8bf*)&sB[crow + 128][ccol] = b2;
        *(v8bf*)&sB[crow + 192][ccol] = b3;
        __syncthreads();
        compute_tile32(sA, sB, acc, wm, wn, l15, akb, bkb);
        __syncthreads();
    }
#endif

    // Epilogue. C/D layout: VGPR r holds M = r + 8*hi, N = l15 (per 16x16 frag).
#pragma unroll
    for (int mf = 0; mf < 4; ++mf) {
#pragma unroll
        for (int nf = 0; nf < 4; ++nf) {
            const size_t gn = tileN + wn + nf * 16 + l15;
            float bv = 0.f;
            if constexpr (EPI != 0) bv = bias[gn];
#pragma unroll
            for (int r = 0; r < 8; ++r) {
                const size_t gm  = tileM + wm + mf * 16 + r + hi * 8;
                const size_t idx = gm * (size_t)ldd + gn;
                float v = acc[mf][nf][r] + bv;
                if constexpr (EPI == 3 || EPI == 4) v = v > 0.f ? v : 0.f;
                if constexpr (EPI == 4) v += resid[bz * batchR + idx];
                if constexpr (EPI == 0 || EPI == 2 || EPI == 4)
                    Df[bz * batchD + idx] = v;
                else
                    Db[bz * batchD + idx] = (bf16_t)v;
            }
        }
    }
}

// ---------------------------------------------------------------------------
// Row-wise LayerNorm over E=1024, 256 threads per row (4 elems/thread).
// TRANS=true writes Y transposed per batch: Y[b][e][s] (feeds GEMM Bt path).
// ---------------------------------------------------------------------------
template <bool TRANS>
__global__ __launch_bounds__(256) void layernorm_kernel(
    const float* __restrict__ X, const float* __restrict__ g,
    const float* __restrict__ b, bf16_t* __restrict__ Y, int S)
{
    __shared__ float redS[8];
    __shared__ float redQ[8];
    const int row = blockIdx.x;           // 0 .. B*S-1
    const int tid = threadIdx.x;
    const float* x = X + (size_t)row * EE;

    float vals[4];
    float lsum = 0.f, lsq = 0.f;
#pragma unroll
    for (int i = 0; i < 4; ++i) {
        const float v = x[tid + i * 256];
        vals[i] = v; lsum += v; lsq += v * v;
    }
#pragma unroll
    for (int off = 16; off > 0; off >>= 1) {
        lsum += __shfl_xor(lsum, off, 32);
        lsq  += __shfl_xor(lsq,  off, 32);
    }
    const int lane = tid & 31, wv = tid >> 5;
    if (lane == 0) { redS[wv] = lsum; redQ[wv] = lsq; }
    __syncthreads();
    float ts = 0.f, tq = 0.f;
#pragma unroll
    for (int i = 0; i < 8; ++i) { ts += redS[i]; tq += redQ[i]; }
    const float mu  = ts * (1.f / EE);
    const float var = tq * (1.f / EE) - mu * mu;
    const float inv = rsqrtf(var + LN_EPS);

    const size_t bb = row / S, s = row % S;
#pragma unroll
    for (int i = 0; i < 4; ++i) {
        const int e = tid + i * 256;
        const float y = (vals[i] - mu) * inv * g[e] + b[e];
        if constexpr (TRANS)
            Y[bb * (size_t)EE * S + (size_t)e * S + s] = (bf16_t)y;
        else
            Y[(size_t)row * EE + e] = (bf16_t)y;
    }
}

// ---------------------------------------------------------------------------
// Masked row softmax over S=2048 (8 elems/thread), scale = 1/sqrt(E).
// ---------------------------------------------------------------------------
__global__ __launch_bounds__(256) void softmax_kernel(
    const float* __restrict__ Sc, const unsigned char* __restrict__ mask,
    bf16_t* __restrict__ P, float scale)
{
    __shared__ float red[8];
    const int row = blockIdx.x;           // b*S + q
    const int tid = threadIdx.x;
    const float* x = Sc + (size_t)row * SS;
    const unsigned char* m = mask + (size_t)(row / SS) * SS;

    float v[8];
    float mx = -INFINITY;
#pragma unroll
    for (int i = 0; i < 8; ++i) {
        const int k = tid + i * 256;
        const float s = m[k] ? x[k] * scale : -INFINITY;
        v[i] = s; mx = fmaxf(mx, s);
    }
#pragma unroll
    for (int off = 16; off > 0; off >>= 1) mx = fmaxf(mx, __shfl_xor(mx, off, 32));
    const int lane = tid & 31, wv = tid >> 5;
    if (lane == 0) red[wv] = mx;
    __syncthreads();
#pragma unroll
    for (int i = 0; i < 8; ++i) mx = fmaxf(mx, red[i]);
    __syncthreads();

    float sum = 0.f;
#pragma unroll
    for (int i = 0; i < 8; ++i) { v[i] = __expf(v[i] - mx); sum += v[i]; }
#pragma unroll
    for (int off = 16; off > 0; off >>= 1) sum += __shfl_xor(sum, off, 32);
    if (lane == 0) red[wv] = sum;
    __syncthreads();
    float tot = 0.f;
#pragma unroll
    for (int i = 0; i < 8; ++i) tot += red[i];
    const float inv = 1.f / tot;
#pragma unroll
    for (int i = 0; i < 8; ++i)
        P[(size_t)row * SS + tid + i * 256] = (bf16_t)(v[i] * inv);
}

// f32 -> bf16 elementwise (4 per thread)
__global__ __launch_bounds__(256) void cvt_bf16_kernel(
    const float* __restrict__ X, bf16_t* __restrict__ Y)
{
    const size_t i = ((size_t)blockIdx.x * 256 + threadIdx.x) * 4;
#pragma unroll
    for (int j = 0; j < 4; ++j) Y[i + j] = (bf16_t)X[i + j];
}

// W[K][N] f32 -> Wt[N][K] bf16  (tiled 32x32 through LDS)
__global__ __launch_bounds__(256) void transpose_cvt_kernel(
    const float* __restrict__ W, bf16_t* __restrict__ Wt, int K, int N)
{
    __shared__ float t[32][33];
    const int kb = blockIdx.y * 32, nb = blockIdx.x * 32;
    const int tx = threadIdx.x & 31, ty = threadIdx.x >> 5;  // 32 x 8
#pragma unroll
    for (int i = 0; i < 32; i += 8)
        t[ty + i][tx] = W[(size_t)(kb + ty + i) * N + nb + tx];
    __syncthreads();
#pragma unroll
    for (int i = 0; i < 32; i += 8)
        Wt[(size_t)(nb + ty + i) * K + kb + tx] = (bf16_t)t[tx][ty + i];
}

// ---------------------------------------------------------------------------
extern "C" void kernel_launch(void* const* d_in, const int* in_sizes, int n_in,
                              void* d_out, int out_size, void* d_ws, size_t ws_size,
                              hipStream_t stream)
{
    const float* x     = (const float*)d_in[0];
    const unsigned char* mask = (const unsigned char*)d_in[1];
    const float* Wq_w  = (const float*)d_in[2];
    const float* Wq_b  = (const float*)d_in[3];
    const float* Wk_w  = (const float*)d_in[4];
    const float* Wk_b  = (const float*)d_in[5];
    const float* Wv_w  = (const float*)d_in[6];
    const float* Wv_b  = (const float*)d_in[7];
    const float* ln1_g = (const float*)d_in[8];
    const float* ln1_b = (const float*)d_in[9];
    const float* ln2_g = (const float*)d_in[10];
    const float* ln2_b = (const float*)d_in[11];
    const float* W1    = (const float*)d_in[12];
    const float* b1    = (const float*)d_in[13];
    const float* W2    = (const float*)d_in[14];
    const float* b2    = (const float*)d_in[15];
    float* out = (float*)d_out;

    const size_t M = (size_t)BB * SS;     // 16384 flattened rows

    // ---- workspace layout (regions overlaid by lifetime) ----
    char* ws = (char*)d_ws;
    size_t off = 0;
    auto alloc = [&](size_t bytes) -> void* {
        void* p = ws + off;
        off += (bytes + 255) & ~(size_t)255;
        return p;
    };
    bf16_t* WqT = (bf16_t*)alloc((size_t)EE * EE * 2);
    bf16_t* WkT = (bf16_t*)alloc((size_t)EE * EE * 2);
    bf16_t* WvT = (bf16_t*)alloc((size_t)EE * EE * 2);
    bf16_t* W1T = (bf16_t*)alloc((size_t)EE * HH * 2);   // Bt: [H][E]
    bf16_t* W2T = (bf16_t*)alloc((size_t)HH * EE * 2);   // Bt: [E][H]
    float*  Vf  = (float*)alloc(M * EE * 4);             // V, kept for residual
    bf16_t* VnT = (bf16_t*)alloc(M * EE * 2);            // LN1(V) transposed per batch
    bf16_t* xb  = (bf16_t*)alloc(M * EE * 2);            // x bf16  (later: h0)
    bf16_t* Qb  = (bf16_t*)alloc(M * EE * 2);            // Q bf16  (later: P low half)
    bf16_t* Kb  = (bf16_t*)alloc(M * EE * 2);            // K bf16  (later: P high half)
    float*  Sc  = (float*)alloc((size_t)BB * SS * SS * 4); // scores (later: ctx, h1)
    if (off > ws_size) return;  // workspace too small; nothing sane to do

    bf16_t* h0  = xb;            // overlay: x dead after QKV
    bf16_t* P   = Qb;            // overlay: Q/K dead after scores (64MB total)
    float*  ctx = Sc;            // overlay: scores dead after softmax
    bf16_t* h1  = (bf16_t*)Sc;   // overlay: ctx dead after LN2

    const dim3 blk(256);

    // 1) weights -> bf16, N-major (Bt) layout
    transpose_cvt_kernel<<<dim3(EE / 32, EE / 32), blk, 0, stream>>>(Wq_w, WqT, EE, EE);
    transpose_cvt_kernel<<<dim3(EE / 32, EE / 32), blk, 0, stream>>>(Wk_w, WkT, EE, EE);
    transpose_cvt_kernel<<<dim3(EE / 32, EE / 32), blk, 0, stream>>>(Wv_w, WvT, EE, EE);
    transpose_cvt_kernel<<<dim3(HH / 32, EE / 32), blk, 0, stream>>>(W1, W1T, EE, HH);
    transpose_cvt_kernel<<<dim3(EE / 32, HH / 32), blk, 0, stream>>>(W2, W2T, HH, EE);
    cvt_bf16_kernel<<<(unsigned)(M * EE / 1024), blk, 0, stream>>>(x, xb);

    // 2) QKV projections: [16384,1024] x [1024,1024]
    gemm_bf16_wmma<1><<<dim3(EE / 256, M / 128, 1), blk, 0, stream>>>(
        xb, EE, 0, WqT, EE, 0, nullptr, Qb, EE, 0, Wq_b, nullptr, 0, EE);
    gemm_bf16_wmma<1><<<dim3(EE / 256, M / 128, 1), blk, 0, stream>>>(
        xb, EE, 0, WkT, EE, 0, nullptr, Kb, EE, 0, Wk_b, nullptr, 0, EE);
    gemm_bf16_wmma<2><<<dim3(EE / 256, M / 128, 1), blk, 0, stream>>>(
        xb, EE, 0, WvT, EE, 0, Vf, nullptr, EE, 0, Wv_b, nullptr, 0, EE);

    // 3) Vn = LN1(V), written transposed per batch -> Bt for attn*V GEMM
    layernorm_kernel<true><<<(unsigned)M, blk, 0, stream>>>(Vf, ln1_g, ln1_b, VnT, SS);

    // 4) scores = Q @ K^T per batch: [2048,1024] x [2048,1024]^T
    gemm_bf16_wmma<0><<<dim3(SS / 256, SS / 128, BB), blk, 0, stream>>>(
        Qb, EE, (size_t)SS * EE, Kb, EE, (size_t)SS * EE,
        Sc, nullptr, SS, (size_t)SS * SS, nullptr, nullptr, 0, EE);

    // 5) P = softmax(mask(scores) / 32)
    softmax_kernel<<<(unsigned)M, blk, 0, stream>>>(Sc, mask, P, 0.03125f);

    // 6) ctx = P @ Vn per batch: [2048,2048] x [2048,1024]
    gemm_bf16_wmma<0><<<dim3(EE / 256, SS / 128, BB), blk, 0, stream>>>(
        P, SS, (size_t)SS * SS, VnT, SS, (size_t)EE * SS,
        ctx, nullptr, EE, (size_t)SS * EE, nullptr, nullptr, 0, SS);

    // 7) h0 = LN2(ctx), row-major bf16
    layernorm_kernel<false><<<(unsigned)M, blk, 0, stream>>>(ctx, ln2_g, ln2_b, h0, SS);

    // 8) h1 = relu(h0 @ W1 + b1): [16384,1024] x [1024,4096]
    gemm_bf16_wmma<3><<<dim3(HH / 256, M / 128, 1), blk, 0, stream>>>(
        h0, EE, 0, W1T, EE, 0, nullptr, h1, HH, 0, b1, nullptr, 0, EE);

    // 9) out = V + relu(h1 @ W2 + b2): [16384,4096] x [4096,1024]
    gemm_bf16_wmma<4><<<dim3(EE / 256, M / 128, 1), blk, 0, stream>>>(
        h1, HH, 0, W2T, HH, 0, out, nullptr, EE, 0, b2, Vf, 0, HH);
}
